// SUnit_23484881174764
// MI455X (gfx1250) — matrix operand order repaired and checked by reading, not compiled, not run on hardware
//
#include <hip/hip_runtime.h>

// ---------------------------------------------------------------------------
// Problem constants
// ---------------------------------------------------------------------------
#define NSZ   256            // image side & DFT size
#define NN    (NSZ * NSZ)    // 65536 elements per plane
#define BATCH 256            // number of images
#define SAMP  128            // sampled sub-grid

typedef unsigned short ushort_t;
typedef __attribute__((ext_vector_type(16))) __bf16 v16bf;
typedef __attribute__((ext_vector_type(8)))  float  v8f;

// ---------------------------------------------------------------------------
// bf16 split-precision helpers (fp32 = hi + lo, both bf16)
// ---------------------------------------------------------------------------
__device__ __forceinline__ ushort_t f2bf(float x) {
  unsigned u = __builtin_bit_cast(unsigned, x);
  unsigned r = u + 0x7fffu + ((u >> 16) & 1u);   // round-to-nearest-even
  return (ushort_t)(r >> 16);
}
__device__ __forceinline__ float bf2f(ushort_t h) {
  unsigned u = ((unsigned)h) << 16;
  return __builtin_bit_cast(float, u);
}

// ---------------------------------------------------------------------------
// Fragment-ready swizzled layouts (waves load fragments as raw b128 pairs).
// A-matrix 16x32 (MxK) bf16 fragment: lane<16 holds M=lane, K=0..15;
//                                     lane>=16 holds M=lane-16, K=16..31
// B-matrix 32x16 (KxN) bf16 fragment: lane<16 holds N=lane,  K=0..15;
//                                     lane>=16 holds N=lane-16, K=16..31
// Storage: [block][lane][e], 512 elements per 16x32 / 32x16 block.
// ---------------------------------------------------------------------------
__device__ __forceinline__ int idxA(int m, int k) {
  const int kk   = k & 31;
  const int half = (kk >> 4) & 1;
  const int e    = kk & 15;
  const int lane = half * 16 + (m & 15);
  return (((m >> 4) * 8) + (k >> 5)) * 512 + lane * 16 + e;
}
__device__ __forceinline__ int idxB(int k, int n) {
  const int kk   = k & 31;
  const int lane = ((kk >> 4) & 1) * 16 + (n & 15);
  const int e    = kk & 15;
  return (((k >> 5) * 16) + (n >> 4)) * 512 + lane * 16 + e;
}

__device__ __forceinline__ v16bf ldfrag(const ushort_t* p) {
  v16bf r;
  __builtin_memcpy(&r, p, sizeof(r));   // 32B aligned -> 2x global_load_b128
  return r;
}

__device__ __forceinline__ void st_split(ushort_t* H, ushort_t* L, long i, float x) {
  const ushort_t h = f2bf(x);
  H[i] = h;
  L[i] = f2bf(x - bf2f(h));
}

__device__ __forceinline__ v8f wmma_acc(v16bf a, v16bf b, v8f c) {
  return __builtin_amdgcn_wmma_f32_16x16x32_bf16(
      false, a, false, b, (short)0, c, false, false);
}
// acc += a*b, split precision: hi*hi + hi*lo + lo*hi
__device__ __forceinline__ v8f fma3(v16bf ah, v16bf al, v16bf bh, v16bf bl, v8f acc) {
  acc = wmma_acc(ah, bh, acc);
  acc = wmma_acc(ah, bl, acc);
  acc = wmma_acc(al, bh, acc);
  return acc;
}

// ---------------------------------------------------------------------------
// Stage 1: fused dwt2 -> shrink -> idwt2(2a - w); local per 2x2 pixel block.
// Output pre-split in A-fragment layout (consumed as A by GEMM 1).
// ---------------------------------------------------------------------------
__device__ __forceinline__ float shrink_g(float t) {
  const float a = fabsf(t) - 1.0f;                 // thresh = 1/BETA = 1
  const float sh = (a > 0.0f) ? copysignf(a, t) : 0.0f;
  return 2.0f * sh - t;                            // 2*shrink(t) - t
}

__global__ void haar_kernel(const float* __restrict__ img,
                            ushort_t* __restrict__ Xh, ushort_t* __restrict__ Xl) {
  const long id = (long)blockIdx.x * blockDim.x + threadIdx.x; // BATCH*128*128
  const int j = (int)(id & 127);
  const int i = (int)((id >> 7) & 127);
  const int b = (int)(id >> 14);
  const float* p = img + (long)b * NN;
  const float a0 = p[(2 * i) * NSZ + 2 * j];
  const float b0 = p[(2 * i) * NSZ + 2 * j + 1];
  const float c0 = p[(2 * i + 1) * NSZ + 2 * j];
  const float d0 = p[(2 * i + 1) * NSZ + 2 * j + 1];
  float ll = (a0 + b0 + c0 + d0) * 0.5f;
  float lh = (a0 - b0 + c0 - d0) * 0.5f;
  float hl = (a0 + b0 - c0 - d0) * 0.5f;
  float hh = (a0 - b0 - c0 + d0) * 0.5f;
  ll = shrink_g(ll); lh = shrink_g(lh); hl = shrink_g(hl); hh = shrink_g(hh);
  const long base = (long)b * NN;
  st_split(Xh, Xl, base + idxA(2 * i,     2 * j),     (ll + lh + hl + hh) * 0.5f);
  st_split(Xh, Xl, base + idxA(2 * i,     2 * j + 1), (ll - lh + hl - hh) * 0.5f);
  st_split(Xh, Xl, base + idxA(2 * i + 1, 2 * j),     (ll + lh - hl - hh) * 0.5f);
  st_split(Xh, Xl, base + idxA(2 * i + 1, 2 * j + 1), (ll - lh - hl + hh) * 0.5f);
}

// ---------------------------------------------------------------------------
// DFT matrix W[r][c] = exp(-2*pi*i*r*c/256), pre-split bf16 hi/lo, stored in
// both A- and B-fragment layouts with +imag and -imag (exact conjugates).
// ---------------------------------------------------------------------------
__global__ void wgen_kernel(ushort_t* __restrict__ WBr_h, ushort_t* __restrict__ WBr_l,
                            ushort_t* __restrict__ WBip_h, ushort_t* __restrict__ WBip_l,
                            ushort_t* __restrict__ WBin_h, ushort_t* __restrict__ WBin_l,
                            ushort_t* __restrict__ WAr_h, ushort_t* __restrict__ WAr_l,
                            ushort_t* __restrict__ WAip_h, ushort_t* __restrict__ WAip_l,
                            ushort_t* __restrict__ WAin_h, ushort_t* __restrict__ WAin_l) {
  const int idx = blockIdx.x * blockDim.x + threadIdx.x;      // NN
  const int r = idx >> 8, c = idx & 255;
  const int m = (r * c) & 255;                                // angle mod 2*pi
  const float th = -0.024543692606170260f * (float)m;         // -2*pi/256 * m
  const float wr = cosf(th);
  const float wi = sinf(th);
  const int iB = idxB(r, c);
  const int iA = idxA(r, c);
  st_split(WBr_h, WBr_l, iB, wr);
  st_split(WAr_h, WAr_l, iA, wr);
  const ushort_t h = f2bf(wi);
  const ushort_t l = f2bf(wi - bf2f(h));
  WBip_h[iB] = h;            WBip_l[iB] = l;
  WBin_h[iB] = h ^ 0x8000u;  WBin_l[iB] = l ^ 0x8000u;
  WAip_h[iA] = h;            WAip_l[iA] = l;
  WAin_h[iA] = h ^ 0x8000u;  WAin_l[iA] = l ^ 0x8000u;
}

// ---------------------------------------------------------------------------
// Batched complex GEMM, one wave per 32x32 output tile (2x2 register blocking:
// every fragment feeds 2 sub-tiles -> 48 WMMAs per 32 b128 loads per K-step).
//   C = A * B   (CPLX=false: A real)
//   mode 0/1: store pre-split into consumer layout (outLayoutA: A vs B);
//             mode 1 adds the sampled-position fixup with s
//   mode 2  : interleaved complex64 store to final output with scale
// ---------------------------------------------------------------------------
template <bool CPLX>
__global__ __launch_bounds__(32)
void cgemm_kernel(const ushort_t* __restrict__ aRh, const ushort_t* __restrict__ aRl,
                  const ushort_t* __restrict__ aIh, const ushort_t* __restrict__ aIl,
                  long aStride,
                  const ushort_t* __restrict__ bRh, const ushort_t* __restrict__ bRl,
                  const ushort_t* __restrict__ bIh, const ushort_t* __restrict__ bIl,
                  long bStride,
                  ushort_t* __restrict__ oRh, ushort_t* __restrict__ oRl,
                  ushort_t* __restrict__ oIh, ushort_t* __restrict__ oIl,
                  int outLayoutA,
                  float* __restrict__ outIlv, const float* __restrict__ s,
                  int mode, float scale)
{
  const int img   = blockIdx.z;
  const int tileM = blockIdx.y * 32;
  const int tileN = blockIdx.x * 32;
  const int lane  = threadIdx.x & 31;

  const long aOff = (long)img * aStride;
  const long bOff = (long)img * bStride;
  const ushort_t* pARh = aRh + aOff;
  const ushort_t* pARl = aRl + aOff;
  const ushort_t* pAIh = CPLX ? aIh + aOff : nullptr;
  const ushort_t* pAIl = CPLX ? aIl + aOff : nullptr;
  const ushort_t* pBRh = bRh + bOff;
  const ushort_t* pBRl = bRl + bOff;
  const ushort_t* pBIh = bIh + bOff;
  const ushort_t* pBIl = bIl + bOff;

  // A blocks: mblk stride 8*512 elems; B blocks: nblk stride 512 elems
  int aB = (tileM >> 4) * 8 * 512 + lane * 16;    // +512  per k-block
  int bB = (tileN >> 4) * 512 + lane * 16;        // +8192 per k-block

  v8f rr[2][2] = {}; v8f ii[2][2] = {}; v8f ri[2][2] = {}; v8f ir[2][2] = {};

#pragma unroll
  for (int kb = 0; kb < 8; ++kb) {
    if (kb < 7) {  // pull next K panel toward the WGP (global_prefetch_b8)
      __builtin_prefetch(pARh + aB + 512, 0, 1);
      __builtin_prefetch(pBRh + bB + 8192, 0, 1);
    }
    v16bf ARh[2], ARl[2], AIh[2], AIl[2];
    v16bf BRh[2], BRl[2], BIh[2], BIl[2];
#pragma unroll
    for (int mi = 0; mi < 2; ++mi) {
      ARh[mi] = ldfrag(pARh + aB + mi * 4096);
      ARl[mi] = ldfrag(pARl + aB + mi * 4096);
      if (CPLX) {
        AIh[mi] = ldfrag(pAIh + aB + mi * 4096);
        AIl[mi] = ldfrag(pAIl + aB + mi * 4096);
      }
    }
#pragma unroll
    for (int ni = 0; ni < 2; ++ni) {
      BRh[ni] = ldfrag(pBRh + bB + ni * 512);
      BRl[ni] = ldfrag(pBRl + bB + ni * 512);
      BIh[ni] = ldfrag(pBIh + bB + ni * 512);
      BIl[ni] = ldfrag(pBIl + bB + ni * 512);
    }
#pragma unroll
    for (int mi = 0; mi < 2; ++mi) {
#pragma unroll
      for (int ni = 0; ni < 2; ++ni) {
        rr[mi][ni] = fma3(ARh[mi], ARl[mi], BRh[ni], BRl[ni], rr[mi][ni]);
        ri[mi][ni] = fma3(ARh[mi], ARl[mi], BIh[ni], BIl[ni], ri[mi][ni]);
        if (CPLX) {
          ii[mi][ni] = fma3(AIh[mi], AIl[mi], BIh[ni], BIl[ni], ii[mi][ni]);
          ir[mi][ni] = fma3(AIh[mi], AIl[mi], BRh[ni], BRl[ni], ir[mi][ni]);
        }
      }
    }
    aB += 512;
    bB += 16 * 512;
  }

  // C layout: VGPR v -> row sub-tile + v + 8*(lane>=16); col sub-tile + lane&15
  const int half = lane >> 4;
#pragma unroll
  for (int mi = 0; mi < 2; ++mi) {
#pragma unroll
    for (int ni = 0; ni < 2; ++ni) {
      const int cN = tileN + ni * 16 + (lane & 15);
#pragma unroll
      for (int v = 0; v < 8; ++v) {
        const int r = tileM + mi * 16 + v + half * 8;
        float cre = rr[mi][ni][v] - ii[mi][ni][v];
        float cim = ri[mi][ni][v] + ir[mi][ni][v];
        if (mode == 1) {
          if ((r & 1) && (cN & 1)) {                 // sampled = odd x odd
            const int si = ((r - 3) & 255) >> 1;
            const int sj = ((cN - 3) & 255) >> 1;
            const float sv = s[((long)img * SAMP + si) * SAMP + sj];
            cre = 1.05f * cre + 0.0525f * sv;        // (1+c)F + c(1+c)s
            cim = 1.05f * cim;
          }
        }
        if (mode == 2) {
          const long o = ((long)img * NN + (long)r * NSZ + cN) * 2;
          outIlv[o]     = cre * scale;               // complex64 interleaved
          outIlv[o + 1] = cim * scale;
        } else {
          const long o = (long)img * NN + (outLayoutA ? idxA(r, cN) : idxB(r, cN));
          st_split(oRh, oRl, o, cre);
          st_split(oIh, oIl, o, cim);
        }
      }
    }
  }
}

// ---------------------------------------------------------------------------
// Launch: out = ifft2( fft2(idwtx) with sampled fixup ), via W-matrix GEMMs
// ---------------------------------------------------------------------------
extern "C" void kernel_launch(void* const* d_in, const int* in_sizes, int n_in,
                              void* d_out, int out_size, void* d_ws, size_t ws_size,
                              hipStream_t stream)
{
  const float* s   = (const float*)d_in[0];   // (256,128,128)
  const float* img = (const float*)d_in[1];   // (256,256,256)

  char* ws = (char*)d_ws;
  size_t off = 0;
  auto alloc16 = [&](size_t elems) {
    ushort_t* p = (ushort_t*)(ws + off);
    off += (elems * sizeof(ushort_t) + 255) & ~(size_t)255;
    return p;
  };

  // W tables: 12 planes of NN bf16 each (~1.5 MiB total, L2-resident)
  ushort_t* WBr_h  = alloc16(NN); ushort_t* WBr_l  = alloc16(NN);
  ushort_t* WBip_h = alloc16(NN); ushort_t* WBip_l = alloc16(NN);
  ushort_t* WBin_h = alloc16(NN); ushort_t* WBin_l = alloc16(NN);
  ushort_t* WAr_h  = alloc16(NN); ushort_t* WAr_l  = alloc16(NN);
  ushort_t* WAip_h = alloc16(NN); ushort_t* WAip_l = alloc16(NN);
  ushort_t* WAin_h = alloc16(NN); ushort_t* WAin_l = alloc16(NN);

  // 8 reusable batched bf16 planes (BATCH*NN each, 32 MiB per plane)
  const size_t PEL = (size_t)BATCH * NN;
  ushort_t* P0 = alloc16(PEL); ushort_t* P1 = alloc16(PEL);
  ushort_t* P2 = alloc16(PEL); ushort_t* P3 = alloc16(PEL);
  ushort_t* P4 = alloc16(PEL); ushort_t* P5 = alloc16(PEL);
  ushort_t* P6 = alloc16(PEL); ushort_t* P7 = alloc16(PEL);

  // X (A-layout, real): P0,P1. U (B-layout, complex): P2..P5.
  // Y (A-layout, complex): P0,P1 (X dead) + P6,P7. V (B-layout): P2..P5 (U dead).
  ushort_t *Xh = P0, *Xl = P1;
  ushort_t *URh = P2, *URl = P3, *UIh = P4, *UIl = P5;
  ushort_t *YRh = P0, *YRl = P1, *YIh = P6, *YIl = P7;
  ushort_t *VRh = P2, *VRl = P3, *VIh = P4, *VIl = P5;

  wgen_kernel<<<NN / 256, 256, 0, stream>>>(WBr_h, WBr_l, WBip_h, WBip_l,
                                            WBin_h, WBin_l, WAr_h, WAr_l,
                                            WAip_h, WAip_l, WAin_h, WAin_l);
  haar_kernel<<<(BATCH * SAMP * SAMP) / 256, 256, 0, stream>>>(img, Xh, Xl);

  dim3 grid(NSZ / 32, NSZ / 32, BATCH), blk(32);

  // U = X * W                  (A real, per-image; B = W shared)
  cgemm_kernel<false><<<grid, blk, 0, stream>>>(
      Xh, Xl, nullptr, nullptr, (long)NN,
      WBr_h, WBr_l, WBip_h, WBip_l, 0L,
      URh, URl, UIh, UIl, /*outLayoutA=*/0,
      nullptr, nullptr, 0, 1.0f);
  // Y = W * U  + sampled fixup
  cgemm_kernel<true><<<grid, blk, 0, stream>>>(
      WAr_h, WAr_l, WAip_h, WAip_l, 0L,
      URh, URl, UIh, UIl, (long)NN,
      YRh, YRl, YIh, YIl, /*outLayoutA=*/1,
      nullptr, s, 1, 1.0f);
  // V = Y * conj(W)
  cgemm_kernel<true><<<grid, blk, 0, stream>>>(
      YRh, YRl, YIh, YIl, (long)NN,
      WBr_h, WBr_l, WBin_h, WBin_l, 0L,
      VRh, VRl, VIh, VIl, /*outLayoutA=*/0,
      nullptr, nullptr, 0, 1.0f);
  // out = conj(W) * V / 256^2   (interleaved complex64)
  cgemm_kernel<true><<<grid, blk, 0, stream>>>(
      WAr_h, WAr_l, WAin_h, WAin_l, 0L,
      VRh, VRl, VIh, VIl, (long)NN,
      nullptr, nullptr, nullptr, nullptr, 0,
      (float*)d_out, nullptr, 2, 1.0f / 65536.0f);
}
